// YambdaTransformer_67010079752832
// MI455X (gfx1250) — compile-verified, matrix operand order checked by reading, not run on hardware
//
#include <hip/hip_runtime.h>
#include <hip/hip_bf16.h>
#include <math.h>

// ---------------------------------------------------------------------------
// YambdaTransformer forward on MI455X (gfx1250, wave32).
// All matmuls run through v_wmma_f32_16x16x32_bf16 (f32 accumulate).
// Block tile 128x64x32, 8 waves, each wave 32x32 (2x2 WMMA register block).
// ---------------------------------------------------------------------------

typedef __attribute__((ext_vector_type(16))) __bf16 v16bf;
typedef __attribute__((ext_vector_type(8)))  float  v8f;

__device__ __forceinline__ unsigned short f32_to_bf16_rne(float f) {
  unsigned int u = __float_as_uint(f);
  u += 0x7FFFu + ((u >> 16) & 1u);   // round to nearest even
  return (unsigned short)(u >> 16);
}
__device__ __forceinline__ unsigned int pack2(unsigned short lo, unsigned short hi) {
  return (unsigned int)lo | ((unsigned int)hi << 16);
}
__device__ __forceinline__ uint4 cvt8_pack(float4 a, float4 b) {
  uint4 p;
  p.x = pack2(f32_to_bf16_rne(a.x), f32_to_bf16_rne(a.y));
  p.y = pack2(f32_to_bf16_rne(a.z), f32_to_bf16_rne(a.w));
  p.z = pack2(f32_to_bf16_rne(b.x), f32_to_bf16_rne(b.y));
  p.w = pack2(f32_to_bf16_rne(b.z), f32_to_bf16_rne(b.w));
  return p;
}

#define TM 128
#define TN 64
#define TK 32

__global__ __launch_bounds__(256) void gemm_bf16_wmma(
    const float* __restrict__ A,  long long sAb, long long sAh, int lda,
    const float* __restrict__ Bm, long long sBb, long long sBh, int ldb, int transB,
    float* __restrict__ C,        long long sCb, long long sCh, int ldc,
    int M, int N, int K, int batchH,
    const float* __restrict__ bias,                 // [N] or null
    const float* __restrict__ resid, int ldr,       // residual add, or null
    const float* __restrict__ pe,   int peMod,      // post-act add pe[(m%peMod)*N+n]
    const float* __restrict__ orgb,                 // per (z,m) bias, or null
    const float* __restrict__ mask,                 // padding mask (B,S): mask[bb*N+n]
    const float* __restrict__ mulp, int ldmul,      // post-act elementwise multiplier
    int act,                                        // 0 none, 1 relu, 2 sigmoid
    float alpha)
{
  __shared__ __align__(16) unsigned short As[TM][TK];
  __shared__ __align__(16) unsigned short Bs[TN][TK];   // stored transposed: [n][k]

  const int z  = blockIdx.z;
  const int bb = z / batchH;
  const int hh = z % batchH;
  const float* Ab = A  + (long long)bb * sAb + (long long)hh * sAh;
  const float* Bb = Bm + (long long)bb * sBb + (long long)hh * sBh;
  float*       Cb = C  + (long long)bb * sCb + (long long)hh * sCh;

  const int row0 = blockIdx.y * TM;
  const int col0 = blockIdx.x * TN;

  const int t    = threadIdx.x;
  const int lane = t & 31;
  const int wave = t >> 5;
  const int wr   = wave >> 1;   // 0..3 -> 32-row slab
  const int wc   = wave & 1;    // 0..1 -> 32-col slab

  const bool ldaAl = ((lda & 3) == 0);   // rows 16B-composable (fails only K=513 gate)
  const bool ldbAl = ((ldb & 3) == 0);

  v8f acc[2][2] = {};

  for (int k0 = 0; k0 < K; k0 += TK) {
    const bool kFull = (k0 + TK <= K);
    // ---- stage A tile (128x32): thread t -> two 8-element segments ----
#pragma unroll
    for (int sseg = 0; sseg < 2; ++sseg) {
      const int e  = t + sseg * 256;
      const int r  = e >> 2;
      const int kk = (e & 3) * 8;
      const int gr = row0 + r;
      const float* src = Ab + (size_t)gr * lda + k0 + kk;
      uint4 pk;
      if (gr < M && kFull && ldaAl) {
        float4 x0 = *(const float4*)(src);
        float4 x1 = *(const float4*)(src + 4);
        pk = cvt8_pack(x0, x1);
      } else {
        unsigned short h[8];
#pragma unroll
        for (int j = 0; j < 8; ++j) {
          const int gk = k0 + kk + j;
          h[j] = f32_to_bf16_rne((gr < M && gk < K) ? src[j] : 0.0f);
        }
        pk.x = pack2(h[0], h[1]); pk.y = pack2(h[2], h[3]);
        pk.z = pack2(h[4], h[5]); pk.w = pack2(h[6], h[7]);
      }
      *(uint4*)&As[r][kk] = pk;
      if (k0 + TK < K)   // speculative prefetch of next K-tile
        __builtin_prefetch(src + TK, 0, 1);
    }
    // ---- stage B tile transposed into Bs[n][k] ----
    if (transB) {        // memory layout (N x K): contiguous in k
      const int n  = t >> 2;
      const int kk = (t & 3) * 8;
      const int gn = col0 + n;
      const float* src = Bb + (size_t)gn * ldb + k0 + kk;
      uint4 pk;
      if (gn < N && kFull && ldbAl) {
        float4 x0 = *(const float4*)(src);
        float4 x1 = *(const float4*)(src + 4);
        pk = cvt8_pack(x0, x1);
      } else {
        unsigned short h[8];
#pragma unroll
        for (int j = 0; j < 8; ++j) {
          const int gk = k0 + kk + j;
          h[j] = f32_to_bf16_rne((gn < N && gk < K) ? src[j] : 0.0f);
        }
        pk.x = pack2(h[0], h[1]); pk.y = pack2(h[2], h[3]);
        pk.z = pack2(h[4], h[5]); pk.w = pack2(h[6], h[7]);
      }
      *(uint4*)&Bs[n][kk] = pk;
    } else {             // memory layout (K x N): contiguous in n -> transpose in LDS
      const int k  = t >> 3;
      const int nn = (t & 7) * 8;
      const int gk = k0 + k;
      const float* src = Bb + (size_t)gk * ldb + col0 + nn;
      unsigned short h[8];
      if (kFull && (col0 + TN <= N) && ldbAl) {
        float4 x0 = *(const float4*)(src);
        float4 x1 = *(const float4*)(src + 4);
        h[0] = f32_to_bf16_rne(x0.x); h[1] = f32_to_bf16_rne(x0.y);
        h[2] = f32_to_bf16_rne(x0.z); h[3] = f32_to_bf16_rne(x0.w);
        h[4] = f32_to_bf16_rne(x1.x); h[5] = f32_to_bf16_rne(x1.y);
        h[6] = f32_to_bf16_rne(x1.z); h[7] = f32_to_bf16_rne(x1.w);
      } else {
#pragma unroll
        for (int j = 0; j < 8; ++j) {
          const int gn = col0 + nn + j;
          h[j] = f32_to_bf16_rne((gk < K && gn < N) ? src[j] : 0.0f);
        }
      }
#pragma unroll
      for (int j = 0; j < 8; ++j) Bs[nn + j][k] = h[j];
    }
    __syncthreads();

    // ---- fragments per the CDNA5 16-bit operand layout ----
    // lane l: row/col = l&15; kb = (l>>4)*8 -> K in [kb,kb+8) (elems 0..7)
    // and [kb+16,kb+24) (elems 8..15).
    const int kb = (lane >> 4) * 8;
    v16bf af[2], bf[2];
#pragma unroll
    for (int ms = 0; ms < 2; ++ms) {
      const int rr = wr * 32 + ms * 16 + (lane & 15);
      __builtin_memcpy(&af[ms],              &As[rr][kb],      16);
      __builtin_memcpy((char*)&af[ms] + 16,  &As[rr][kb + 16], 16);
    }
#pragma unroll
    for (int ns = 0; ns < 2; ++ns) {
      const int cc = wc * 32 + ns * 16 + (lane & 15);
      __builtin_memcpy(&bf[ns],              &Bs[cc][kb],      16);
      __builtin_memcpy((char*)&bf[ns] + 16,  &Bs[cc][kb + 16], 16);
    }
#pragma unroll
    for (int ms = 0; ms < 2; ++ms)
#pragma unroll
      for (int ns = 0; ns < 2; ++ns)
        acc[ms][ns] = __builtin_amdgcn_wmma_f32_16x16x32_bf16(
            false, af[ms], false, bf[ns], (short)0, acc[ms][ns], false, false);
    __syncthreads();
  }

  // ---- epilogue: C/D layout — lane l: n = l&15; VGPR v: m = v + 8*(l>>4) ----
#pragma unroll
  for (int ms = 0; ms < 2; ++ms) {
    const int mBase = row0 + wr * 32 + ms * 16 + ((lane >> 4) << 3);
#pragma unroll
    for (int ns = 0; ns < 2; ++ns) {
      const int nBase = col0 + wc * 32 + ns * 16 + (lane & 15);
#pragma unroll
      for (int v = 0; v < 8; ++v) {
        const int mrow = mBase + v;
        const int ncol = nBase;
        if (mrow < M && ncol < N) {
          float val = acc[ms][ns][v] * alpha;
          if (bias)  val += bias[ncol];
          if (resid) val += resid[(size_t)mrow * ldr + ncol];
          if (orgb)  val += orgb[(size_t)z * M + mrow];
          if (mask && mask[(size_t)bb * N + ncol] == 0.0f) val = -1e9f;
          if      (act == 1) val = val > 0.0f ? val : 0.0f;
          else if (act == 2) val = 1.0f / (1.0f + __expf(-val));
          if (pe)    val += pe[(size_t)(mrow % peMod) * N + ncol];
          if (mulp)  val *= mulp[(size_t)mrow * ldmul + ncol];
          Cb[(size_t)mrow * ldc + ncol] = val;
        }
      }
    }
  }
}

// ---------------------------------------------------------------------------
// Row softmax (in-place). One block per row.
// ---------------------------------------------------------------------------
__global__ __launch_bounds__(256) void softmax_rows(float* __restrict__ x, int ncol) {
  __shared__ float red[256];
  float* p = x + (size_t)blockIdx.x * ncol;
  const int t = threadIdx.x;
  float mx = -3.4e38f;
  for (int c = t; c < ncol; c += 256) mx = fmaxf(mx, p[c]);
  red[t] = mx; __syncthreads();
  for (int o = 128; o > 0; o >>= 1) { if (t < o) red[t] = fmaxf(red[t], red[t + o]); __syncthreads(); }
  mx = red[0]; __syncthreads();
  float sum = 0.0f;
  for (int c = t; c < ncol; c += 256) { float e = __expf(p[c] - mx); p[c] = e; sum += e; }
  red[t] = sum; __syncthreads();
  for (int o = 128; o > 0; o >>= 1) { if (t < o) red[t] += red[t + o]; __syncthreads(); }
  const float inv = 1.0f / red[0];
  for (int c = t; c < ncol; c += 256) p[c] *= inv;
}

// ---------------------------------------------------------------------------
// Row LayerNorm: out[r] = (in[r]-mean)*rsqrt(var+eps)*s + b
// ---------------------------------------------------------------------------
__global__ __launch_bounds__(256) void layernorm_rows(
    const float* __restrict__ in, int ldin,
    float* __restrict__ out, int ldout,
    const float* __restrict__ s, const float* __restrict__ b, int ncol)
{
  __shared__ float red[256];
  const float* p = in + (size_t)blockIdx.x * ldin;
  float*       q = out + (size_t)blockIdx.x * ldout;
  const int t = threadIdx.x;
  float sum = 0.0f;
  for (int c = t; c < ncol; c += 256) sum += p[c];
  red[t] = sum; __syncthreads();
  for (int o = 128; o > 0; o >>= 1) { if (t < o) red[t] += red[t + o]; __syncthreads(); }
  const float mean = red[0] / (float)ncol; __syncthreads();
  float vs = 0.0f;
  for (int c = t; c < ncol; c += 256) { float d = p[c] - mean; vs += d * d; }
  red[t] = vs; __syncthreads();
  for (int o = 128; o > 0; o >>= 1) { if (t < o) red[t] += red[t + o]; __syncthreads(); }
  const float rstd = rsqrtf(red[0] / (float)ncol + 1e-5f);
  for (int c = t; c < ncol; c += 256) q[c] = (p[c] - mean) * rstd * s[c] + b[c];
}

// Embedding gather with padding_idx=0 (row 0 forced to zero).
__global__ void gather_emb(const int* __restrict__ ids, const float* __restrict__ tbl,
                           float* __restrict__ out, int ldout, int Dd, int n) {
  int i = blockIdx.x * blockDim.x + threadIdx.x;
  if (i >= n * Dd) return;
  int r = i / Dd, c = i - r * Dd;
  int id = ids[r];
  out[(size_t)r * ldout + c] = (id == 0) ? 0.0f : tbl[(size_t)id * Dd + c];
}

// orgb[b,h,s] = is_organic[b,s]*worg[h] + borg[h]
__global__ void org_bias(const float* __restrict__ is_org, const float* __restrict__ worg,
                         const float* __restrict__ borg, float* __restrict__ out,
                         int Bn, int Hn, int Sn) {
  int i = blockIdx.x * blockDim.x + threadIdx.x;
  if (i >= Bn * Hn * Sn) return;
  int s = i % Sn, h = (i / Sn) % Hn, b = i / (Sn * Hn);
  out[i] = is_org[b * Sn + s] * worg[h] + borg[h];
}

// Fill column D (=512) of the gate-concat buffer with is_organic.
__global__ void fill_org_col(const float* __restrict__ is_org, float* __restrict__ Xg,
                             int ld, int col, int n) {
  int i = blockIdx.x * blockDim.x + threadIdx.x;
  if (i < n) Xg[(size_t)i * ld + col] = is_org[i];
}

// ---------------------------------------------------------------------------
// Host orchestration
// ---------------------------------------------------------------------------
static inline void launch_gemm(hipStream_t st,
    const float* A, long long sAb, long long sAh, int lda,
    const float* B, long long sBb, long long sBh, int ldb, int transB,
    float* C, long long sCb, long long sCh, int ldc,
    int M, int N, int K, int batch, int batchH,
    const float* bias, const float* resid, int ldr,
    const float* pe, int peMod,
    const float* orgb, const float* mask,
    const float* mulp, int ldmul, int act, float alpha)
{
  dim3 grid((unsigned)((N + TN - 1) / TN), (unsigned)((M + TM - 1) / TM), (unsigned)batch);
  gemm_bf16_wmma<<<grid, 256, 0, st>>>(A, sAb, sAh, lda, B, sBb, sBh, ldb, transB,
                                       C, sCb, sCh, ldc, M, N, K, batchH,
                                       bias, resid, ldr, pe, peMod, orgb, mask,
                                       mulp, ldmul, act, alpha);
}

extern "C" void kernel_launch(void* const* d_in, const int* in_sizes, int n_in,
                              void* d_out, int out_size, void* d_ws, size_t ws_size,
                              hipStream_t stream) {
  constexpr int Bc = 8, Sc = 512, Dc = 512, Hc = 8, Lc = 4, DFFc = 2048, NIc = 20000, ADc = 128;
  constexpr int Mr = Bc * Sc;            // 4096 rows
  const float scale = 0.125f;            // 1/sqrt(DK=64)

  const int*   item_ids = (const int*)  d_in[0];
  const float* audio    = (const float*)d_in[1];
  const float* is_org   = (const float*)d_in[2];
  const float* pmask    = (const float*)d_in[3];   // (B,1,1,S) == (B,S)
  const float* embt     = (const float*)d_in[4];
  const float* pe       = (const float*)d_in[5];
  const float* Wa1 = (const float*)d_in[6];  const float* ba1 = (const float*)d_in[7];
  const float* Wa2 = (const float*)d_in[8];  const float* ba2 = (const float*)d_in[9];
  const float* Wf  = (const float*)d_in[10]; const float* bfb = (const float*)d_in[11];
  const float* Wq  = (const float*)d_in[12]; const float* bq  = (const float*)d_in[13];
  const float* Wk  = (const float*)d_in[14]; const float* bk  = (const float*)d_in[15];
  const float* Wv  = (const float*)d_in[16]; const float* bv  = (const float*)d_in[17];
  const float* Wo  = (const float*)d_in[18]; const float* bo  = (const float*)d_in[19];
  const float* worg= (const float*)d_in[20]; const float* borg= (const float*)d_in[21];
  const float* ln1s= (const float*)d_in[22]; const float* ln1b= (const float*)d_in[23];
  const float* ln2s= (const float*)d_in[24]; const float* ln2b= (const float*)d_in[25];
  const float* Wff1= (const float*)d_in[26]; const float* bff1= (const float*)d_in[27];
  const float* Wff2= (const float*)d_in[28]; const float* bff2= (const float*)d_in[29];
  const float* lnOs= (const float*)d_in[30]; const float* lnOb= (const float*)d_in[31];
  const float* Wg  = (const float*)d_in[32]; const float* bg  = (const float*)d_in[33];
  const float* Wout= (const float*)d_in[34]; const float* bout= (const float*)d_in[35];
  (void)in_sizes; (void)n_in; (void)out_size; (void)ws_size;

  // ---- workspace carve (~195 MB of f32) ----
  float* w = (float*)d_ws;
  float* Xc     = w; w += (size_t)Mr * (2 * Dc);     // concat [item | audio_emb]
  float* Ya     = w; w += (size_t)Mr * Dc;           // audio hidden
  float* x      = w; w += (size_t)Mr * Dc;           // residual stream
  float* tmp    = w; w += (size_t)Mr * Dc;           // pre-LN sums
  float* qb     = w; w += (size_t)Mr * Dc;
  float* kb2    = w; w += (size_t)Mr * Dc;
  float* vb     = w; w += (size_t)Mr * Dc;
  float* ctx    = w; w += (size_t)Mr * Dc;
  float* ffh    = w; w += (size_t)Mr * DFFc;
  float* scores = w; w += (size_t)Bc * Hc * Sc * Sc; // 64 MB
  float* orgb   = w; w += (size_t)Bc * Hc * Sc;
  float* Xg     = w; w += (size_t)Mr * (Dc + 1);     // [LN(x) | is_organic]
  float* xg     = w; w += (size_t)Mr * Dc;           // gated features

  // ---- 1. item embedding gather -> Xc[:, 0:512] ----
  {
    int n = Mr * Dc;
    gather_emb<<<(n + 255) / 256, 256, 0, stream>>>(item_ids, embt, Xc, 2 * Dc, Dc, Mr);
  }
  // ---- 2. audio MLP ----
  launch_gemm(stream, audio, 0, 0, ADc, Wa1, 0, 0, Dc, 0, Ya, 0, 0, Dc,
              Mr, Dc, ADc, 1, 1, ba1, nullptr, 0, nullptr, 1, nullptr, nullptr,
              nullptr, 0, /*relu*/1, 1.0f);
  launch_gemm(stream, Ya, 0, 0, Dc, Wa2, 0, 0, Dc, 0, Xc + Dc, 0, 0, 2 * Dc,
              Mr, Dc, Dc, 1, 1, ba2, nullptr, 0, nullptr, 1, nullptr, nullptr,
              nullptr, 0, 0, 1.0f);
  // ---- 3. fuse: x = relu(Xc @ Wf + bf) + pe ----
  launch_gemm(stream, Xc, 0, 0, 2 * Dc, Wf, 0, 0, Dc, 0, x, 0, 0, Dc,
              Mr, Dc, 2 * Dc, 1, 1, bfb, nullptr, 0, pe, Sc, nullptr, nullptr,
              nullptr, 0, /*relu*/1, 1.0f);

  // ---- 4. transformer layers ----
  for (int l = 0; l < Lc; ++l) {
    const float* Wql = Wq + (size_t)l * Dc * Dc;   const float* bql = bq + (size_t)l * Dc;
    const float* Wkl = Wk + (size_t)l * Dc * Dc;   const float* bkl = bk + (size_t)l * Dc;
    const float* Wvl = Wv + (size_t)l * Dc * Dc;   const float* bvl = bv + (size_t)l * Dc;
    const float* Wol = Wo + (size_t)l * Dc * Dc;   const float* bol = bo + (size_t)l * Dc;
    const float* W1l = Wff1 + (size_t)l * Dc * DFFc; const float* b1l = bff1 + (size_t)l * DFFc;
    const float* W2l = Wff2 + (size_t)l * DFFc * Dc; const float* b2l = bff2 + (size_t)l * Dc;

    {
      int n = Bc * Hc * Sc;
      org_bias<<<(n + 255) / 256, 256, 0, stream>>>(is_org, worg + l * Hc, borg + l * Hc,
                                                    orgb, Bc, Hc, Sc);
    }
    // QKV projections
    launch_gemm(stream, x, 0, 0, Dc, Wql, 0, 0, Dc, 0, qb, 0, 0, Dc, Mr, Dc, Dc, 1, 1,
                bql, nullptr, 0, nullptr, 1, nullptr, nullptr, nullptr, 0, 0, 1.0f);
    launch_gemm(stream, x, 0, 0, Dc, Wkl, 0, 0, Dc, 0, kb2, 0, 0, Dc, Mr, Dc, Dc, 1, 1,
                bkl, nullptr, 0, nullptr, 1, nullptr, nullptr, nullptr, 0, 0, 1.0f);
    launch_gemm(stream, x, 0, 0, Dc, Wvl, 0, 0, Dc, 0, vb, 0, 0, Dc, Mr, Dc, Dc, 1, 1,
                bvl, nullptr, 0, nullptr, 1, nullptr, nullptr, nullptr, 0, 0, 1.0f);
    // scores[b,h] = scale * Q[b,h] @ K[b,h]^T  + orgbias (per query) + mask
    launch_gemm(stream,
                qb,  (long long)Sc * Dc, 64, Dc,                   // A: (S x 64) per head
                kb2, (long long)Sc * Dc, 64, Dc, /*transB=*/1,     // B: (S x 64) read as N x K
                scores, (long long)Hc * Sc * Sc, (long long)Sc * Sc, Sc,
                Sc, Sc, 64, Bc * Hc, Hc,
                nullptr, nullptr, 0, nullptr, 1, orgb, pmask, nullptr, 0, 0, scale);
    softmax_rows<<<Bc * Hc * Sc, 256, 0, stream>>>(scores, Sc);
    // ctx[b,h] = P[b,h] @ V[b,h]
    launch_gemm(stream,
                scores, (long long)Hc * Sc * Sc, (long long)Sc * Sc, Sc,
                vb, (long long)Sc * Dc, 64, Dc, 0,
                ctx, (long long)Sc * Dc, 64, Dc,
                Sc, 64, Sc, Bc * Hc, Hc,
                nullptr, nullptr, 0, nullptr, 1, nullptr, nullptr, nullptr, 0, 0, 1.0f);
    // out proj + residual -> tmp; LN -> x
    launch_gemm(stream, ctx, 0, 0, Dc, Wol, 0, 0, Dc, 0, tmp, 0, 0, Dc, Mr, Dc, Dc, 1, 1,
                bol, x, Dc, nullptr, 1, nullptr, nullptr, nullptr, 0, 0, 1.0f);
    layernorm_rows<<<Mr, 256, 0, stream>>>(tmp, Dc, x, Dc, ln1s + l * Dc, ln1b + l * Dc, Dc);
    // FFN
    launch_gemm(stream, x, 0, 0, Dc, W1l, 0, 0, DFFc, 0, ffh, 0, 0, DFFc,
                Mr, DFFc, Dc, 1, 1, b1l, nullptr, 0, nullptr, 1, nullptr, nullptr,
                nullptr, 0, /*relu*/1, 1.0f);
    launch_gemm(stream, ffh, 0, 0, DFFc, W2l, 0, 0, Dc, 0, tmp, 0, 0, Dc,
                Mr, Dc, DFFc, 1, 1, b2l, x, Dc, nullptr, 1, nullptr, nullptr,
                nullptr, 0, 0, 1.0f);
    layernorm_rows<<<Mr, 256, 0, stream>>>(tmp, Dc, x, Dc, ln2s + l * Dc, ln2b + l * Dc, Dc);
  }

  // ---- 5. final LN into gate-concat buffer, append is_organic column ----
  layernorm_rows<<<Mr, 256, 0, stream>>>(x, Dc, Xg, Dc + 1, lnOs, lnOb, Dc);
  fill_org_col<<<(Mr + 255) / 256, 256, 0, stream>>>(is_org, Xg, Dc + 1, Dc, Mr);

  // ---- 6. gate: xg = sigmoid(Xg @ Wg + bg) * LN(x)   (K = 513, guarded) ----
  launch_gemm(stream, Xg, 0, 0, Dc + 1, Wg, 0, 0, Dc, 0, xg, 0, 0, Dc,
              Mr, Dc, Dc + 1, 1, 1, bg, nullptr, 0, nullptr, 1, nullptr, nullptr,
              Xg, Dc + 1, /*sigmoid*/2, 1.0f);

  // ---- 7. output head: logits = xg @ Wout + bout  -> d_out (4096 x 20000) ----
  launch_gemm(stream, xg, 0, 0, Dc, Wout, 0, 0, NIc, 0, (float*)d_out, 0, 0, NIc,
              Mr, NIc, Dc, 1, 1, bout, nullptr, 0, nullptr, 1, nullptr, nullptr,
              nullptr, 0, 0, 1.0f);
}
